// QuickDistogram_25881472925811
// MI455X (gfx1250) — compile-verified, matrix operand order checked by reading, not run on hardware
//
#include <hip/hip_runtime.h>

typedef __attribute__((ext_vector_type(16))) _Float16 v16h;
typedef __attribute__((ext_vector_type(8)))  float    v8f;

#define NROWS 32768
#define KNBR  64
#define CFEAT 256
#define PAIRD 32
#define HIDD  64
#define BINS  16
#define CLIPV 32

static __device__ __forceinline__ v8f wmma_f16(v16h a, v16h b, v8f c) {
  // D = A(16x32 f16) * B(32x16 f16) + C(16x16 f32)
  return __builtin_amdgcn_wmma_f32_16x16x32_f16(false, a, false, b, (short)0, c,
                                                false, false);
}

// gelu(x) = 0.5x(1+tanh(0.79788456(x+0.044715x^3)))
// tanh(u) = 1 - 2/(exp(2u)+1)  -> one v_exp_f32 + one v_rcp_f32.
static __device__ __forceinline__ float gelu_fast(float x) {
  float u = 0.7978845608028654f * (x + 0.044715f * x * x * x);
  float e = __expf(2.0f * u);
  float t = 1.0f - 2.0f / (e + 1.0f);
  return 0.5f * x * (1.0f + t);
}

// Load the 16 fragment dims {0..7, 16..23} (base already includes +off) as
// four aligned float4s -> v[16].
static __device__ __forceinline__ void ld16(const float* __restrict__ p,
                                            float* __restrict__ v) {
  const float4* f4 = (const float4*)p;
  float4 a = f4[0], b = f4[1], c = f4[4], d = f4[5];
  v[0] = a.x;  v[1] = a.y;  v[2]  = a.z;  v[3]  = a.w;
  v[4] = b.x;  v[5] = b.y;  v[6]  = b.z;  v[7]  = b.w;
  v[8] = c.x;  v[9] = c.y;  v[10] = c.z;  v[11] = c.w;
  v[12] = d.x; v[13] = d.y; v[14] = d.z;  v[15] = d.w;
}

static __device__ __forceinline__ v16h cvt16(const float* __restrict__ v) {
  v16h a;
#pragma unroll
  for (int i = 0; i < 16; ++i) a[i] = (_Float16)v[i];
  return a;
}

// ---------------------------------------------------------------------------
// Kernel 0: pack weights into per-lane WMMA B-fragment layout (f16).
//   projB: 32 fragments (kc 0..7) x (Wl c0, Wl c1, Wr c0, Wr c1)
//   w1f  : 4 fragments  (hidden column blocks)
//   w2f  : 2 fragments  (K blocks of W2)
// Fragment element (frag, lane, i): lane<16 -> K {0..7,16..23}, lane>=16 -> +8.
// ---------------------------------------------------------------------------
__global__ __launch_bounds__(256) void qd_pack(
    const float* __restrict__ Wl, const float* __restrict__ Wr,
    const float* __restrict__ W1, const float* __restrict__ W2,
    _Float16* __restrict__ projB, _Float16* __restrict__ w1f,
    _Float16* __restrict__ w2f)
{
  const int tid = blockIdx.x * blockDim.x + threadIdx.x;
  const int nthr = blockDim.x * gridDim.x;

  for (int idx = tid; idx < 32 * 32 * 16; idx += nthr) {
    int i = idx & 15, lane = (idx >> 4) & 31, frag = idx >> 9;
    int kc = frag >> 2, m = frag & 3;
    int laneLo = lane & 15, hi = lane >> 4;
    int K = kc * 32 + ((i < 8) ? i : (i + 8)) + (hi ? 8 : 0);
    int col = (m & 1) * 16 + laneLo;
    const float* W = (m < 2) ? Wl : Wr;
    projB[idx] = (_Float16)W[K * PAIRD + col];
  }
  for (int idx = tid; idx < 4 * 32 * 16; idx += nthr) {
    int i = idx & 15, lane = (idx >> 4) & 31, c = idx >> 9;
    int laneLo = lane & 15, hi = lane >> 4;
    int K = ((i < 8) ? i : (i + 8)) + (hi ? 8 : 0);
    w1f[idx] = (_Float16)W1[K * HIDD + c * 16 + laneLo];
  }
  for (int idx = tid; idx < 2 * 32 * 16; idx += nthr) {
    int i = idx & 15, lane = (idx >> 4) & 31, q = idx >> 9;
    int laneLo = lane & 15, hi = lane >> 4;
    int u = q * 32 + ((i < 8) ? i : (i + 8)) + (hi ? 8 : 0);
    w2f[idx] = (_Float16)W2[u * BINS + laneLo];
  }
}

// ---------------------------------------------------------------------------
// Kernel 1: left = feat @ W_left, right = feat @ W_right.
// One wave per 16 feature rows; A from 4x b128 loads per K-chunk, B fragments
// from packed workspace (2x b128 each) -> 32 v_wmma per wave.
// ---------------------------------------------------------------------------
__global__ __launch_bounds__(256) void qd_proj(
    const float* __restrict__ feat,
    const _Float16* __restrict__ projB,
    float* __restrict__ left,
    float* __restrict__ right)
{
  const int lane   = threadIdx.x & 31;
  const int laneLo = lane & 15;
  const int hi     = lane >> 4;
  const int tile   = blockIdx.x * (blockDim.x >> 5) + (threadIdx.x >> 5);
  const int row0   = tile * 16;

  const v16h* PB = (const v16h*)projB;
  const float* frow = feat + (size_t)(row0 + laneLo) * CFEAT + (hi ? 8 : 0);

  v8f aL0 = {}, aL1 = {}, aR0 = {}, aR1 = {};
#pragma unroll
  for (int kc = 0; kc < 8; ++kc) {
    float av[16];
    ld16(frow + kc * 32, av);
    v16h a = cvt16(av);
    v16h bl0 = PB[(kc * 4 + 0) * 32 + lane];
    v16h bl1 = PB[(kc * 4 + 1) * 32 + lane];
    v16h br0 = PB[(kc * 4 + 2) * 32 + lane];
    v16h br1 = PB[(kc * 4 + 3) * 32 + lane];
    aL0 = wmma_f16(a, bl0, aL0);
    aL1 = wmma_f16(a, bl1, aL1);
    aR0 = wmma_f16(a, br0, aR0);
    aR1 = wmma_f16(a, br1, aR1);
  }
  // D layout: VGPR r, lane -> row = r + 8*hi, col = 16c + laneLo.
#pragma unroll
  for (int r = 0; r < 8; ++r) {
    int row = row0 + r + 8 * hi;
    left [(size_t)row * PAIRD + laneLo]      = aL0[r];
    left [(size_t)row * PAIRD + 16 + laneLo] = aL1[r];
    right[(size_t)row * PAIRD + laneLo]      = aR0[r];
    right[(size_t)row * PAIRD + 16 + laneLo] = aR1[r];
  }
}

// ---------------------------------------------------------------------------
// Kernel 2: per-edge pipeline, one wave per 16 edges sharing source row n.
// dcode -> LayerNorm -> GEMM1 (4 WMMA, C=0, bias folded into GELU) -> LDS
// repack -> GEMM2 (2 chained WMMA) -> log-softmax + expected distance.
// ---------------------------------------------------------------------------
__global__ __launch_bounds__(256) void qd_edges(
    const int*   __restrict__ resi,
    const int*   __restrict__ chain,
    const int*   __restrict__ batch,
    const int*   __restrict__ nbrs,
    const float* __restrict__ Wpos,
    const float* __restrict__ lnS,
    const float* __restrict__ lnB,
    const float* __restrict__ b1,
    const float* __restrict__ b2,
    const _Float16* __restrict__ w1f,
    const _Float16* __restrict__ w2f,
    const float* __restrict__ left,
    const float* __restrict__ right,
    float* __restrict__ logp,
    float* __restrict__ dmap)
{
  __shared__ float lh[8][16 * 72];   // per-wave 16x64 hidden tile, padded rows

  const int lane   = threadIdx.x & 31;
  const int laneLo = lane & 15;
  const int hi     = lane >> 4;
  const int off    = hi ? 8 : 0;
  const int wave   = threadIdx.x >> 5;
  const int tile   = blockIdx.x * (blockDim.x >> 5) + wave;
  const int n      = tile >> 2;
  const int k0     = (tile & 3) << 4;
  const int e0     = n * KNBR + k0;

  const int j = nbrs[e0 + laneLo];
  int rel = resi[j] - resi[n];
  rel = min(max(rel, -CLIPV), CLIPV) + CLIPV;
  const float pm = ((chain[j] == chain[n]) && (batch[j] == batch[n])) ? 1.f : 0.f;

  float lv[16], rv[16], pv[16], sv[16], bv[16];
  ld16(left  + (size_t)n   * PAIRD + off, lv);
  ld16(right + (size_t)j   * PAIRD + off, rv);
  ld16(Wpos  + (size_t)rel * PAIRD + off, pv);
  ld16(lnS + off, sv);
  ld16(lnB + off, bv);

  // dcode (16 of 32 dims per lane) + LayerNorm stats.
  float v[16];
  float s1 = 0.f, s2 = 0.f;
#pragma unroll
  for (int i = 0; i < 16; ++i) {
    float x = lv[i] + rv[i] + pm * pv[i];
    v[i] = x;
    s1 += x;
    s2 += x * x;
  }
  s1 += __shfl_xor(s1, 16, 32);      // partner half-lane = other 16 dims
  s2 += __shfl_xor(s2, 16, 32);
  const float mu  = s1 * (1.0f / 32.0f);
  const float var = s2 * (1.0f / 32.0f) - mu * mu;
  const float inv = rsqrtf(var + 1e-5f);

  v16h a;
#pragma unroll
  for (int i = 0; i < 16; ++i)
    a[i] = (_Float16)((v[i] - mu) * inv * sv[i] + bv[i]);

  // GEMM1: [16 x 32] @ W1[32 x 64], C = inline 0; b1 folded into GELU input.
  const v16h* W1F = (const v16h*)w1f;
  const v8f zero = {};
  v8f h[4];
#pragma unroll
  for (int c = 0; c < 4; ++c)
    h[c] = wmma_f16(a, W1F[c * 32 + lane], zero);

  // GELU in f32 accumulators, scatter D-layout -> A-layout through LDS.
#pragma unroll
  for (int c = 0; c < 4; ++c) {
    float bias = b1[c * 16 + laneLo];
#pragma unroll
    for (int r = 0; r < 8; ++r)
      lh[wave][(r + 8 * hi) * 72 + c * 16 + laneLo] = gelu_fast(h[c][r] + bias);
  }

  __syncthreads();

  // GEMM2: [16 x 64] @ W2[64 x 16] + b2, two chained K=32 WMMAs.
  const v16h* W2F = (const v16h*)w2f;
  float bias2 = b2[laneLo];
  v8f acc;
#pragma unroll
  for (int r = 0; r < 8; ++r) acc[r] = bias2;
  const float* hrow = &lh[wave][laneLo * 72];
#pragma unroll
  for (int q = 0; q < 2; ++q) {
    float a2v[16];
    ld16(hrow + q * 32 + off, a2v);   // 4x ds_load_b128
    v16h a2 = cvt16(a2v);
    acc = wmma_f16(a2, W2F[q * 32 + lane], acc);
  }

  // logits: VGPR r, lane -> edge = r + 8*hi, bin = laneLo.
  const float center = ((float)laneLo + 0.5f) * (22.0f / BINS);
#pragma unroll
  for (int r = 0; r < 8; ++r) {
    float x = acc[r];
    float m = x;
#pragma unroll
    for (int msk = 1; msk < 16; msk <<= 1)
      m = fmaxf(m, __shfl_xor(m, msk, 32));
    float e = __expf(x - m);
    float s = e;
#pragma unroll
    for (int msk = 1; msk < 16; msk <<= 1)
      s += __shfl_xor(s, msk, 32);
    float lse = m + __logf(s);
    float lp  = x - lse;
    int edge = e0 + r + 8 * hi;
    logp[(size_t)edge * BINS + laneLo] = lp;
    float w = __expf(lp) * center;
#pragma unroll
    for (int msk = 1; msk < 16; msk <<= 1)
      w += __shfl_xor(w, msk, 32);
    if (laneLo == 0) dmap[edge] = w;
  }
}

extern "C" void kernel_launch(void* const* d_in, const int* in_sizes, int n_in,
                              void* d_out, int out_size, void* d_ws, size_t ws_size,
                              hipStream_t stream) {
  const float* feat = (const float*)d_in[0];
  const int*   resi = (const int*)  d_in[1];
  const int*   chn  = (const int*)  d_in[2];
  const int*   bat  = (const int*)  d_in[3];
  const int*   nbr  = (const int*)  d_in[4];
  const float* Wl   = (const float*)d_in[5];
  const float* Wr   = (const float*)d_in[6];
  const float* Wpos = (const float*)d_in[7];
  const float* lnS  = (const float*)d_in[8];
  const float* lnB  = (const float*)d_in[9];
  const float* W1   = (const float*)d_in[10];
  const float* b1   = (const float*)d_in[11];
  const float* W2   = (const float*)d_in[12];
  const float* b2   = (const float*)d_in[13];

  float* left  = (float*)d_ws;                           // [N,32] f32
  float* right = left + (size_t)NROWS * PAIRD;           // [N,32] f32
  _Float16* projB = (_Float16*)(right + (size_t)NROWS * PAIRD); // 32*32*16 f16
  _Float16* w1fp  = projB + 32 * 32 * 16;                // 4*32*16 f16
  _Float16* w2fp  = w1fp + 4 * 32 * 16;                  // 2*32*16 f16

  float* logp = (float*)d_out;                           // [N,K,16]
  float* dm   = logp + (size_t)NROWS * KNBR * BINS;      // [N,K]

  qd_pack<<<16, 256, 0, stream>>>(Wl, Wr, W1, W2, projB, w1fp, w2fp);
  qd_proj<<<NROWS / 16 / 8, 256, 0, stream>>>(feat, projB, left, right);
  qd_edges<<<(NROWS * KNBR) / 16 / 8, 256, 0, stream>>>(
      resi, chn, bat, nbr, Wpos, lnS, lnB, b1, b2, w1fp, w2fp,
      left, right, logp, dm);
}
